// GraphNet_74088185856643
// MI455X (gfx1250) — matrix operand order repaired and checked by reading, not compiled, not run on hardware
//
#include <hip/hip_runtime.h>

#define NN   16384
#define EE   524288
#define HID1 200
#define H1P  208   // 13 x 16 tiles
#define HID2 8
#define BK   32
#define BM   128

typedef __bf16 bf16_t;
typedef bf16_t v16bf __attribute__((ext_vector_type(16)));
typedef float  v8f   __attribute__((ext_vector_type(8)));
typedef int    v4i   __attribute__((vector_size(16)));   // matches builtin param type

struct U128 { unsigned int a, b, c, d; };
struct U256 { U128 lo, hi; };

#define AS1 __attribute__((address_space(1)))
#define AS3 __attribute__((address_space(3)))

#if defined(__gfx1250__) && __has_builtin(__builtin_amdgcn_global_load_async_to_lds_b128)
#define HAVE_ASYNC_LDS 1
#else
#define HAVE_ASYNC_LDS 0
#endif

// 16-byte global -> LDS copy; async (ASYNCcnt-tracked) when the builtin exists.
__device__ __forceinline__ void copy16_g2s(const void* g, void* s) {
#if HAVE_ASYNC_LDS
  __builtin_amdgcn_global_load_async_to_lds_b128(
      (AS1 v4i*)(size_t)g, (AS3 v4i*)(unsigned int)(size_t)s, 0, 0);
#else
  *(U128*)s = *(const U128*)g;
#endif
}

__device__ __forceinline__ void wait_async_lds() {
#if HAVE_ASYNC_LDS
#if __has_builtin(__builtin_amdgcn_s_wait_asynccnt)
  __builtin_amdgcn_s_wait_asynccnt(0);
#else
  asm volatile("s_wait_asynccnt 0" ::: "memory");
#endif
#endif
}

// Split fp32 into bf16 hi + bf16 lo (truncating), x ~= hi + lo to ~16 mantissa bits.
__device__ __forceinline__ void split2(float v, unsigned short& h, unsigned short& l) {
  unsigned int bits = __float_as_uint(v);
  h = (unsigned short)(bits >> 16);
  float hf = __uint_as_float(((unsigned int)h) << 16);
  l = (unsigned short)(__float_as_uint(v - hf) >> 16);
}

// ---------------- degree / norm ----------------
__global__ void k_zero(float* p, int n) {
  int i = blockIdx.x * blockDim.x + threadIdx.x;
  if (i < n) p[i] = 0.0f;
}

__global__ void k_deg(const int* __restrict__ col, const float* __restrict__ wgt,
                      float* __restrict__ deg) {
  int e = blockIdx.x * blockDim.x + threadIdx.x;
  if (e < EE) atomicAdd(&deg[col[e]], wgt[e]);
}

__global__ void k_dinv(const float* __restrict__ deg, float* __restrict__ dinv) {
  int i = blockIdx.x * blockDim.x + threadIdx.x;
  if (i < NN) {
    float d = deg[i] + 1.0f;         // self-loop weight 1.0
    dinv[i] = 1.0f / sqrtf(d);       // d >= 1 always
  }
}

// ---------------- one-time W1 -> transposed bf16 hi/lo planes ----------------
// W1T_*[n][k]: n in 0..207 (cols 200..207 zero), k in 0..16383; coalesced reads,
// 64B-chunk writes via LDS transpose.
__global__ void __launch_bounds__(256) k_prepw(const float* __restrict__ W1,
                                               unsigned short* __restrict__ W1T_hi,
                                               unsigned short* __restrict__ W1T_lo) {
  __shared__ alignas(16) unsigned short TH[H1P][32];
  __shared__ alignas(16) unsigned short TL[H1P][32];
  const int tid = threadIdx.x;
  const int k0 = blockIdx.x * 32;
#pragma unroll
  for (int it = 0; it < 26; ++it) {
    int i = tid + 256 * it;        // 32*208 = 6656
    int r = i / H1P;               // k within chunk
    int c = i - r * H1P;           // column n
    float v = (c < HID1) ? W1[(size_t)(k0 + r) * HID1 + c] : 0.0f;
    unsigned short h, l;
    split2(v, h, l);
    TH[c][r] = h; TL[c][r] = l;
  }
  __syncthreads();
#pragma unroll
  for (int it = 0; it < 2; ++it) {
    int j = tid + 256 * it;        // 416 row-copies of 64B
    if (j < 416) {
      int sel = j >= H1P;
      int n = j - sel * H1P;
      const U128* s = (const U128*)&(sel ? TL : TH)[n][0];
      U128* d = (U128*)&(sel ? W1T_lo : W1T_hi)[(size_t)n * NN + k0];
#pragma unroll
      for (int q = 0; q < 4; ++q) d[q] = s[q];
    }
  }
}

// ---------------- layer-1 GEMM: h1 = x @ W1 (bf16x3 split, WMMA) ----------------
__global__ void __launch_bounds__(256) k_gemm1(const float* __restrict__ x,
                                               const unsigned short* __restrict__ W1T_hi,
                                               const unsigned short* __restrict__ W1T_lo,
                                               float* __restrict__ h1) {
  __shared__ alignas(16) unsigned short AsHi[BM][40];   // [row][k], 80B rows
  __shared__ alignas(16) unsigned short AsLo[BM][40];
  __shared__ alignas(16) unsigned short BsHi[H1P][40];  // [n][k], 64B data + pad
  __shared__ alignas(16) unsigned short BsLo[H1P][40];

  const int tid  = threadIdx.x;
  const int lane = tid & 31;
  const int wv   = tid >> 5;
  const int row0 = blockIdx.x * BM;
  const int hf   = lane >> 4;
  const int l16  = lane & 15;

  v8f acc[13];
  v8f zero8 = {0.f, 0.f, 0.f, 0.f, 0.f, 0.f, 0.f, 0.f};
#pragma unroll
  for (int t = 0; t < 13; ++t) acc[t] = zero8;

#pragma unroll 1
  for (int k0 = 0; k0 < NN; k0 += BK) {
    // --- B tile: pure 16B copies (async when available), 1664 U128s total ---
#pragma unroll
    for (int it = 0; it < 7; ++it) {
      int i = tid + 256 * it;
      if (i < 1664) {
        int sel = i >= 832;          // 0 = hi plane, 1 = lo plane
        int ii = i - sel * 832;
        int n  = ii >> 2;            // B row (output column)
        int q  = ii & 3;             // 16B chunk within the 64B row
        const U128* src = (const U128*)(sel ? W1T_lo : W1T_hi) +
                          ((size_t)n * (NN / 8) + (k0 >> 3) + q);
        void* dst = &(sel ? BsLo : BsHi)[n][q * 8];
        copy16_g2s(src, dst);
      }
    }
    // --- A tile: 128x32 fp32 -> bf16 hi/lo, float4-coalesced ---
#pragma unroll
    for (int it = 0; it < 4; ++it) {
      int r = (tid >> 3) + it * 32;
      int c = (tid & 7) * 4;
      const float4 v = *(const float4*)&x[(size_t)(row0 + r) * NN + k0 + c];
      unsigned short h, l;
      split2(v.x, h, l); AsHi[r][c + 0] = h; AsLo[r][c + 0] = l;
      split2(v.y, h, l); AsHi[r][c + 1] = h; AsLo[r][c + 1] = l;
      split2(v.z, h, l); AsHi[r][c + 2] = h; AsLo[r][c + 2] = l;
      split2(v.w, h, l); AsHi[r][c + 3] = h; AsLo[r][c + 3] = l;
    }
    wait_async_lds();
    __syncthreads();

    // --- A fragments (lanes<16: K{0..7,16..23}; lanes>=16: K{8..15,24..31}) ---
    const int rowA = wv * 16 + l16;
    U256 tah, tal;
    {
      const U128* p = (const U128*)&AsHi[rowA][hf * 8];
      tah.lo = p[0]; tah.hi = p[2];
      const U128* q = (const U128*)&AsLo[rowA][hf * 8];
      tal.lo = q[0]; tal.hi = q[2];
    }
    v16bf aH = __builtin_bit_cast(v16bf, tah);
    v16bf aL = __builtin_bit_cast(v16bf, tal);

#pragma unroll
    for (int t = 0; t < 13; ++t) {
      U256 tbh, tbl;
      const U128* p = (const U128*)&BsHi[t * 16 + l16][hf * 8];
      tbh.lo = p[0]; tbh.hi = p[2];
      const U128* q = (const U128*)&BsLo[t * 16 + l16][hf * 8];
      tbl.lo = q[0]; tbl.hi = q[2];
      v16bf bH = __builtin_bit_cast(v16bf, tbh);
      v16bf bL = __builtin_bit_cast(v16bf, tbl);
      // bf16x3: hi*hi + hi*lo + lo*hi  (lo*lo dropped, ~2^-16 rel err)
      acc[t] = __builtin_amdgcn_wmma_f32_16x16x32_bf16(false, aH, false, bH, (short)0, acc[t], false, false);
      acc[t] = __builtin_amdgcn_wmma_f32_16x16x32_bf16(false, aH, false, bL, (short)0, acc[t], false, false);
      acc[t] = __builtin_amdgcn_wmma_f32_16x16x32_bf16(false, aL, false, bH, (short)0, acc[t], false, false);
    }
    __syncthreads();
  }

  // store: C layout — VGPR v: row = v (lanes 0-15) / v+8 (lanes 16-31), col = lane%16
  const int rbase = row0 + wv * 16 + hf * 8;
#pragma unroll
  for (int t = 0; t < 13; ++t) {
#pragma unroll
    for (int vv = 0; vv < 8; ++vv) {
      h1[(size_t)(rbase + vv) * H1P + t * 16 + l16] = acc[t][vv];
    }
  }
}

// ---------------- layer-1 aggregation ----------------
__global__ void k_agg1_init(const float* __restrict__ h1, const float* __restrict__ dinv,
                            const float* __restrict__ b1, float* __restrict__ a1) {
  int g = blockIdx.x * blockDim.x + threadIdx.x;
  if (g >= NN * HID1) return;
  int n = g / HID1;
  int j = g - n * HID1;
  float di = dinv[n];
  a1[(size_t)n * H1P + j] = di * di * h1[(size_t)n * H1P + j] + b1[j];
}

__global__ void k_agg1_edge(const int* __restrict__ ei, const float* __restrict__ wgt,
                            const float* __restrict__ dinv, const float* __restrict__ h1,
                            float* __restrict__ a1) {
  int e = blockIdx.x;
  int j = threadIdx.x;
  int r = ei[e];
  int c = ei[EE + e];
  float nrm = dinv[r] * wgt[e] * dinv[c];
  if (j < HID1) atomicAdd(&a1[(size_t)c * H1P + j], nrm * h1[(size_t)r * H1P + j]);
}

// ---------------- layer-2 GEMM: h2 = relu(a1) @ W2 (tiny, fp32 scalar) ----------------
__global__ void k_gemm2(const float* __restrict__ a1, const float* __restrict__ W2,
                        float* __restrict__ h2) {
  int g = blockIdx.x * blockDim.x + threadIdx.x;
  if (g >= NN * HID2) return;
  int n = g >> 3;
  int j = g & 7;
  const float* row = &a1[(size_t)n * H1P];
  float acc = 0.0f;
  for (int k = 0; k < HID1; ++k) {
    float v = row[k];
    v = v > 0.0f ? v : 0.0f;  // ReLU fused
    acc += v * W2[k * HID2 + j];
  }
  h2[g] = acc;
}

// ---------------- layer-2 aggregation ----------------
__global__ void k_agg2_init(const float* __restrict__ h2, const float* __restrict__ dinv,
                            const float* __restrict__ b2, float* __restrict__ out) {
  int g = blockIdx.x * blockDim.x + threadIdx.x;
  if (g >= NN * HID2) return;
  int n = g >> 3;
  int j = g & 7;
  float di = dinv[n];
  out[g] = di * di * h2[g] + b2[j];
}

__global__ void k_agg2_edge(const int* __restrict__ ei, const float* __restrict__ wgt,
                            const float* __restrict__ dinv, const float* __restrict__ h2,
                            float* __restrict__ out) {
  int g = blockIdx.x * blockDim.x + threadIdx.x;
  if (g >= EE * HID2) return;
  int e = g >> 3;
  int j = g & 7;
  int r = ei[e];
  int c = ei[EE + e];
  float nrm = dinv[r] * wgt[e] * dinv[c];
  atomicAdd(&out[c * HID2 + j], nrm * h2[r * HID2 + j]);
}

extern "C" void kernel_launch(void* const* d_in, const int* in_sizes, int n_in,
                              void* d_out, int out_size, void* d_ws, size_t ws_size,
                              hipStream_t stream) {
  const float* x   = (const float*)d_in[0];
  const int*   ei  = (const int*)  d_in[1];   // [2, E]: rows then cols
  const float* wgt = (const float*)d_in[2];
  const float* W1  = (const float*)d_in[3];
  const float* b1  = (const float*)d_in[4];
  const float* W2  = (const float*)d_in[5];
  const float* b2  = (const float*)d_in[6];
  float* out = (float*)d_out;

  float* ws   = (float*)d_ws;
  float* deg  = ws;                            // N
  float* dinv = deg + NN;                      // N
  float* h1   = dinv + NN;                     // N * H1P
  float* a1   = h1 + (size_t)NN * H1P;         // N * H1P
  float* h2   = a1 + (size_t)NN * H1P;         // N * H2
  unsigned short* W1T_hi = (unsigned short*)(h2 + (size_t)NN * HID2);  // H1P * N bf16
  unsigned short* W1T_lo = W1T_hi + (size_t)H1P * NN;                  // H1P * N bf16

  k_zero<<<NN / 256, 256, 0, stream>>>(deg, NN);
  k_deg<<<EE / 256, 256, 0, stream>>>(ei + EE, wgt, deg);
  k_dinv<<<NN / 256, 256, 0, stream>>>(deg, dinv);

  k_prepw<<<NN / 32, 256, 0, stream>>>(W1, W1T_hi, W1T_lo);
  k_gemm1<<<NN / BM, 256, 0, stream>>>(x, W1T_hi, W1T_lo, h1);

  k_agg1_init<<<(NN * HID1 + 255) / 256, 256, 0, stream>>>(h1, dinv, b1, a1);
  k_agg1_edge<<<EE, 224, 0, stream>>>(ei, wgt, dinv, h1, a1);

  k_gemm2<<<(NN * HID2) / 256, 256, 0, stream>>>(a1, W2, h2);

  k_agg2_init<<<(NN * HID2) / 256, 256, 0, stream>>>(h2, dinv, b2, out);
  k_agg2_edge<<<(EE * HID2) / 256, 256, 0, stream>>>(ei, wgt, dinv, h2, out);
}
